// LstmCellRudder_22316650070363
// MI455X (gfx1250) — matrix operand order repaired and checked by reading, not compiled, not run on hardware
//
#include <hip/hip_runtime.h>
#include <hip/hip_bf16.h>
#include <math.h>

typedef _Float16 v8h  __attribute__((ext_vector_type(8)));
typedef _Float16 v16h __attribute__((ext_vector_type(16)));
typedef float    v8f  __attribute__((ext_vector_type(8)));

#define BATCH  256
#define TSTEPS 512
#define NPOS   128
#define NACT   64
#define NIN    192
#define HID    512
#define G4H    2048

#define MROWS  16              // batch rows per workgroup
#define NWAVES 16
#define THREADS (NWAVES * 32)  // 512 threads = 16 wave32

// ---------------------------------------------------------------------------
// f32 -> f16 weight conversion (one-time prep into d_ws)
// ---------------------------------------------------------------------------
__global__ void cvt_f32_to_f16(const float* __restrict__ src,
                               _Float16* __restrict__ dst, int n) {
  int i = blockIdx.x * blockDim.x + threadIdx.x;
  if (i < n) dst[i] = (_Float16)src[i];
}

__device__ __forceinline__ float sigmoidf_(float x) {
  return 1.0f / (1.0f + __expf(-x));
}

// ---------------------------------------------------------------------------
// Persistent LSTM kernel: each workgroup owns 16 batch rows for all T steps.
// Per step: gates[16,2048] = bias + x_t[16,192]@W_ih^T + h[16,512]@W_hh^T
// via v_wmma_f32_16x16x32_f16, then per-lane cell update, h -> LDS.
// Wave w owns hidden columns j in [32w, 32w+32) and computes the i/f/g/o
// tiles for exactly those columns, so the cell update needs no gate exchange.
//
// Weight fragments are re-streamed from the L2-resident f16 copy every step.
// `tbase` (value 0) is made opaque by an asm barrier each iteration: this
// stops LICM from hoisting (then spilling) the loop-invariant fragments,
// while keeping the weight POINTERS pristine so the loads lower to
// global_load_b128 with scalar-base addressing (not flat).
// ---------------------------------------------------------------------------
__global__ __launch_bounds__(THREADS, 1)
void lstm_persistent(
    const float* __restrict__ obs,    // [B,T,128]
    const float* __restrict__ act,    // [B,T,64]
    const float* __restrict__ h0,     // [B,H]
    const float* __restrict__ c0,     // [B,H]
    const float* __restrict__ b_ih,   // [4H]
    const float* __restrict__ b_hh,   // [4H]
    const float* __restrict__ w_out,  // [H]
    const float* __restrict__ b_out,  // [1]
    const _Float16* __restrict__ whh16, // [4H][H]  row-major f16
    const _Float16* __restrict__ wih16, // [4H][I]  row-major f16
    float* __restrict__ out)          // [B,T]
{
  __shared__ __attribute__((aligned(32))) _Float16 hbuf[MROWS][HID];
  __shared__ __attribute__((aligned(32))) _Float16 xbuf[MROWS][NIN];
  __shared__ float ledger[MROWS];

  const int tid  = threadIdx.x;
  const int lane = tid & 31;
  const int wv   = tid >> 5;     // wave 0..15
  const int ln   = lane & 15;    // N column within tile / A row
  const int hi   = lane >> 4;    // 0 or 1 (lane half)
  const int b0   = blockIdx.x * MROWS;

  // Per-lane tile constants. gs = g*2 + s: gate g in {i,f,g,o}, sub-tile s.
  unsigned whhOff[8], wihOff[8];
  float biasv[8];
#pragma unroll
  for (int gs = 0; gs < 8; ++gs) {
    int g = gs >> 1, s = gs & 1;
    int col = g * HID + wv * 32 + s * 16 + ln;   // gate column in [0,4H)
    whhOff[gs] = (unsigned)col * HID + (unsigned)hi * 16;  // halves
    wihOff[gs] = (unsigned)col * NIN + (unsigned)hi * 16;  // halves
    biasv[gs]  = b_ih[col] + b_hh[col];
  }
  const float wo0 = w_out[wv * 32 + ln];
  const float wo1 = w_out[wv * 32 + 16 + ln];
  const float bo  = b_out[0];

  // c state in registers: creg[s][r] = c[row = r + hi*8][j = 32w + 16s + ln]
  float creg[2][8];
#pragma unroll
  for (int s = 0; s < 2; ++s)
#pragma unroll
    for (int r = 0; r < 8; ++r)
      creg[s][r] = c0[(size_t)(b0 + r + hi * 8) * HID + wv * 32 + s * 16 + ln];

  // h0 -> LDS (f16), x_0 -> LDS (f16), ledger zero
  for (int e = tid; e < MROWS * HID; e += THREADS) {
    int r = e >> 9, j = e & (HID - 1);
    hbuf[r][j] = (_Float16)h0[(size_t)(b0 + r) * HID + j];
  }
  for (int e = tid; e < MROWS * NIN; e += THREADS) {
    int r = e / NIN, p = e - r * NIN;
    float v = (p < NPOS)
                  ? obs[((size_t)(b0 + r) * TSTEPS + 0) * NPOS + p]
                  : act[((size_t)(b0 + r) * TSTEPS + 0) * NACT + (p - NPOS)];
    xbuf[r][p] = (_Float16)v;
  }
  if (tid < MROWS) ledger[tid] = 0.0f;
  __syncthreads();

  const int am   = ln;        // A-matrix row for this lane
  const int aoff = hi * 16;   // A-matrix K sub-offset (lanes 16-31: K+16)
  const _Float16* hrow = &hbuf[am][0];
  const _Float16* xrow = &xbuf[am][0];

#pragma unroll 1
  for (int t = 0; t < TSTEPS; ++t) {
    // Opaque zero offset: defeats hoisting of weight loads out of the t-loop
    // without obscuring the (global) weight pointers themselves.
    unsigned tbase = 0;
    asm volatile("" : "+s"(tbase));

    // ---- GEMM phase: acc = bias + x_t@W_ih^T + h@W_hh^T -----------------
    v8f acc[8];
#pragma unroll
    for (int gs = 0; gs < 8; ++gs) {
      float bv = biasv[gs];
      v8f a = {bv, bv, bv, bv, bv, bv, bv, bv};
      acc[gs] = a;
    }

    // Input projection, K = 192 (6 chunks of 32)
#pragma unroll
    for (int kb = 0; kb < NIN; kb += 32) {
      v8h alo = *(const v8h*)(xrow + kb + aoff);
      v8h ahi = *(const v8h*)(xrow + kb + 8 + aoff);
      v16h afrag = __builtin_shufflevector(alo, ahi, 0, 1, 2, 3, 4, 5, 6, 7,
                                           8, 9, 10, 11, 12, 13, 14, 15);
      v16h bf[8];
#pragma unroll
      for (int gs = 0; gs < 8; ++gs)
        bf[gs] = *(const v16h*)(wih16 + tbase + wihOff[gs] + kb);
#pragma unroll
      for (int gs = 0; gs < 8; ++gs)
        acc[gs] = __builtin_amdgcn_wmma_f32_16x16x32_f16(
            false, afrag, false, bf[gs], (short)0, acc[gs], false, false);
    }
    // Recurrence, K = 512 (16 chunks of 32)
#pragma unroll 4
    for (int kb = 0; kb < HID; kb += 32) {
      v8h alo = *(const v8h*)(hrow + kb + aoff);
      v8h ahi = *(const v8h*)(hrow + kb + 8 + aoff);
      v16h afrag = __builtin_shufflevector(alo, ahi, 0, 1, 2, 3, 4, 5, 6, 7,
                                           8, 9, 10, 11, 12, 13, 14, 15);
      v16h bf[8];
#pragma unroll
      for (int gs = 0; gs < 8; ++gs)
        bf[gs] = *(const v16h*)(whh16 + tbase + whhOff[gs] + kb);
#pragma unroll
      for (int gs = 0; gs < 8; ++gs)
        acc[gs] = __builtin_amdgcn_wmma_f32_16x16x32_f16(
            false, afrag, false, bf[gs], (short)0, acc[gs], false, false);
    }
    __syncthreads();  // all LDS reads of hbuf/xbuf complete

    // ---- Cell update (pure per-lane VALU; gate tiles align by lane) -----
    float rsum[8];
#pragma unroll
    for (int r = 0; r < 8; ++r) rsum[r] = 0.0f;
#pragma unroll
    for (int s = 0; s < 2; ++s) {
      const int j = wv * 32 + s * 16 + ln;
      const float wo = s ? wo1 : wo0;
#pragma unroll
      for (int r = 0; r < 8; ++r) {
        float iv = sigmoidf_(acc[0 + s][r]);
        float fv = sigmoidf_(acc[2 + s][r]);
        float gv = tanhf(acc[4 + s][r]);
        float ov = sigmoidf_(acc[6 + s][r]);
        float c  = fv * creg[s][r] + iv * gv;
        creg[s][r] = c;
        float h = ov * tanhf(c);
        hbuf[r + hi * 8][j] = (_Float16)h;
        rsum[r] += h * wo;
      }
    }
    // Reduce output partials across the 16 lanes of each half (rows split
    // by lane half), then ds_add_f32 into the per-row ledger.
#pragma unroll
    for (int r = 0; r < 8; ++r) {
      float v = rsum[r];
      v += __shfl_xor(v, 1, 32);
      v += __shfl_xor(v, 2, 32);
      v += __shfl_xor(v, 4, 32);
      v += __shfl_xor(v, 8, 32);
      rsum[r] = v;
    }
    if (ln == 0) {
#pragma unroll
      for (int r = 0; r < 8; ++r) atomicAdd(&ledger[r + hi * 8], rsum[r]);
    }
    // Stage next timestep's x tile (xbuf reads finished at barrier above)
    if (t + 1 < TSTEPS) {
      for (int e = tid; e < MROWS * NIN; e += THREADS) {
        int r = e / NIN, p = e - r * NIN;
        float v = (p < NPOS)
                      ? obs[((size_t)(b0 + r) * TSTEPS + (t + 1)) * NPOS + p]
                      : act[((size_t)(b0 + r) * TSTEPS + (t + 1)) * NACT +
                            (p - NPOS)];
        xbuf[r][p] = (_Float16)v;
      }
    }
    __syncthreads();  // hbuf / ledger / xbuf ready

    if (tid < MROWS) {
      out[(size_t)(b0 + tid) * TSTEPS + t] = ledger[tid] + bo;
      ledger[tid] = 0.0f;  // ordered before next step's atomics by barrier
    }
  }
}

// ---------------------------------------------------------------------------
extern "C" void kernel_launch(void* const* d_in, const int* in_sizes, int n_in,
                              void* d_out, int out_size, void* d_ws,
                              size_t ws_size, hipStream_t stream) {
  const float* obs   = (const float*)d_in[0];
  const float* act   = (const float*)d_in[1];
  const float* h0    = (const float*)d_in[2];
  const float* c0    = (const float*)d_in[3];
  const float* W_ih  = (const float*)d_in[4];
  const float* W_hh  = (const float*)d_in[5];
  const float* b_ih  = (const float*)d_in[6];
  const float* b_hh  = (const float*)d_in[7];
  const float* W_out = (const float*)d_in[8];
  const float* b_out = (const float*)d_in[9];
  float* out = (float*)d_out;

  // Scratch: f16 copies of the weight matrices (2 MB + 0.75 MB)
  _Float16* whh16 = (_Float16*)d_ws;
  _Float16* wih16 = whh16 + (size_t)G4H * HID;

  const int nhh = G4H * HID;  // 1048576
  const int nih = G4H * NIN;  // 393216
  cvt_f32_to_f16<<<(nhh + 255) / 256, 256, 0, stream>>>(W_hh, whh16, nhh);
  cvt_f32_to_f16<<<(nih + 255) / 256, 256, 0, stream>>>(W_ih, wih16, nih);

  lstm_persistent<<<BATCH / MROWS, THREADS, 0, stream>>>(
      obs, act, h0, c0, b_ih, b_hh, W_out, b_out, whh16, wih16, out);
}
